// ContextualizedNN_67525475827826
// MI455X (gfx1250) — matrix operand order repaired and checked by reading, not compiled, not run on hardware
//
#include <hip/hip_runtime.h>
#include <math.h>

// ---------------- problem constants (from reference) ----------------
#define HOPS   5
#define TOPK   20
#define D_IN   128
#define D_HID  128
#define D_OUT  64
#define BATCH  256
#define IPU    20
#define NOCC   (BATCH * (1 + IPU))     // 5376 occurrences (items + user items)
#define OCC_PER_WG 8
#define ROWS   (OCC_PER_WG * TOPK)     // 160 rows per workgroup (10 M-tiles, no padding)
#define LDA    136                      // bf16 row stride (128 + 8 pad) -> 272B, 16B aligned
#define LDO    68                       // f32 row stride for layer-2 output reuse

// ---------------- LDS layout (dynamic shared, 140032 bytes) ----------------
#define OFF_W1T 0
#define SZ_W1T  (D_HID * LDA * 2)                 // 34816
#define OFF_W2T (OFF_W1T + SZ_W1T)
#define SZ_W2T  (D_OUT * LDA * 2)                 // 17408
#define OFF_A   (OFF_W2T + SZ_W2T)
#define SZ_A    (ROWS * LDA * 2)                  // 43520
#define OFF_H   (OFF_A + SZ_A)
#define SZ_H    (ROWS * LDA * 2)                  // 43520
#define OFF_B1  (OFF_H + SZ_H)
#define SZ_B1   (D_HID * 4)
#define OFF_B2  (OFF_B1 + SZ_B1)
#define SZ_B2   (D_OUT * 4)
#define SMEM_BYTES (OFF_B2 + SZ_B2)               // 140032 (< 160KB -> 2 blocks/WGP)

typedef __attribute__((ext_vector_type(16))) __bf16 bf16x16;
typedef __attribute__((ext_vector_type(8)))  __bf16 bf16x8;
typedef __attribute__((ext_vector_type(8)))  float  f32x8;
typedef __attribute__((ext_vector_type(4)))  unsigned int u32x4;

// A-operand fragment (16x32): lane(0-15)=row M, lane-halves take
// K {base..+7, base+16..+23} / {base+8..+15, base+24..+31}.
__device__ __forceinline__ bf16x16 load_a_frag(const __bf16* rowbase, int kk, int hl) {
  bf16x16 f;
  const __bf16* p = rowbase + kk * 32 + hl * 8;
  ((u32x4*)&f)[0] = *(const u32x4*)p;          // 8 bf16, 16B aligned
  ((u32x4*)&f)[1] = *(const u32x4*)(p + 16);   // next 8 bf16 (+32B)
  return f;
}

// B-operand fragment (32x16): lane(0-15)=col N, lane-halves take
// K {base..+7, base+16..+23} / {base+8..+15, base+24..+31}.
__device__ __forceinline__ bf16x16 load_b_frag(const __bf16* colbase, int kk, int hl) {
  bf16x16 f;
  const __bf16* p = colbase + kk * 32 + hl * 8;
  ((u32x4*)&f)[0] = *(const u32x4*)p;
  ((u32x4*)&f)[1] = *(const u32x4*)(p + 16);
  return f;
}

// One workgroup = 8 occurrences x 1 hop. 8 waves, wave w owns occurrence w's
// 20 neighbor rows during the gather. GEMMs are computed "flipped"
// (H^T = W^T x A^T): weight fragment is WMMA operand A (lane = out-channel),
// activation row fragment is operand B (lane = data row). The C output then
// has 8 consecutive out-channels per lane -> packed ds_store_b128 epilogues.
__global__ __launch_bounds__(256) void rep_kernel(
    const int*   __restrict__ item_idxs,      // [256]
    const int*   __restrict__ user_item_ids,  // [256*20]
    const int*   __restrict__ neighbor_ids,   // [N_ITEMS][HOPS][TOPK]
    const float* __restrict__ neighbor_scores,// [N_ITEMS][HOPS][TOPK]
    const float* __restrict__ embed,          // [N_ITEMS][128]
    const float* __restrict__ W1g,            // [HOPS][128][128]
    const float* __restrict__ b1g,            // [HOPS][128]
    const float* __restrict__ W2g,            // [HOPS][128][64]
    const float* __restrict__ b2g,            // [HOPS][64]
    float*       __restrict__ rep)            // [NOCC][HOPS][64]
{
  extern __shared__ char smem[];
  __bf16* sW1T = (__bf16*)(smem + OFF_W1T);
  __bf16* sW2T = (__bf16*)(smem + OFF_W2T);
  __bf16* sA   = (__bf16*)(smem + OFF_A);
  __bf16* sH   = (__bf16*)(smem + OFF_H);
  float*  sb1  = (float*)(smem + OFF_B1);
  float*  sb2  = (float*)(smem + OFF_B2);

  const int tid  = threadIdx.x;
  const int wave = tid >> 5;
  const int lane = tid & 31;
  const int hl   = lane >> 4;
  const int l16  = lane & 15;
  const int og   = blockIdx.x;   // occurrence group 0..671
  const int hop  = blockIdx.y;   // 0..4

  // ---- Phase 0a: stage per-hop weights into LDS, transposed to [n][k] bf16 ----
  const float* W1 = W1g + (long)hop * D_IN * D_HID;
  for (int i = tid; i < D_IN * D_HID; i += 256) {
    int k = i >> 7, n = i & 127;                 // coalesced over n
    sW1T[n * LDA + k] = (__bf16)W1[i];
  }
  const float* W2 = W2g + (long)hop * D_HID * D_OUT;
  for (int i = tid; i < D_HID * D_OUT; i += 256) {
    int k = i >> 6, n = i & 63;
    sW2T[n * LDA + k] = (__bf16)W2[i];
  }
  if (tid < D_HID) sb1[tid] = b1g[hop * D_HID + tid];
  if (tid < D_OUT) sb2[tid] = b2g[hop * D_OUT + tid];

  // ---- Phase 0b: gather + score-scale + bf16-convert 160 embedding rows ----
  {
    int occ = og * OCC_PER_WG + wave;
    int idx = (occ < BATCH) ? item_idxs[occ] : user_item_ids[occ - BATCH];
    long nb = ((long)idx * HOPS + hop) * TOPK;
    for (int kn = 0; kn < TOPK; ++kn) {
      int   nid = neighbor_ids[nb + kn];
      float sc  = neighbor_scores[nb + kn];
      const float* erow = embed + (long)nid * D_IN;
      if (kn + 1 < TOPK) {  // warm L2/WGP$ for the next gathered row
        __builtin_prefetch(embed + (long)neighbor_ids[nb + kn + 1] * D_IN, 0, 0);
      }
      const float4 f = ((const float4*)erow)[lane];   // 4 floats / lane
      __bf16* dst = &sA[(wave * TOPK + kn) * LDA + lane * 4];
      dst[0] = (__bf16)(f.x * sc);
      dst[1] = (__bf16)(f.y * sc);
      dst[2] = (__bf16)(f.z * sc);
      dst[3] = (__bf16)(f.w * sc);
    }
  }
  __syncthreads();

  // ---- Phase 1: layer 1  H = relu(A x W1 + b1), computed as H^T tiles ----
  // Wave w owns hid-tile w (8 tiles); W1 fragments + biases preloaded once.
  {
    const int ctile = wave;                       // hid-channel tile 0..7
    bf16x16 wf[4];
    const __bf16* wbase = &sW1T[(ctile * 16 + l16) * LDA];
#pragma unroll
    for (int kk = 0; kk < 4; ++kk) wf[kk] = load_a_frag(wbase, kk, hl);
    f32x8 bias;
#pragma unroll
    for (int i = 0; i < 8; ++i) bias[i] = sb1[ctile * 16 + hl * 8 + i];

    for (int rt = 0; rt < 10; ++rt) {             // 10 row-tiles of 16
      f32x8 acc = {};
      const __bf16* abase = &sA[(rt * 16 + l16) * LDA];
#pragma unroll
      for (int kk = 0; kk < 4; ++kk) {
        bf16x16 xf = load_b_frag(abase, kk, hl);
        acc = __builtin_amdgcn_wmma_f32_16x16x32_bf16(
            false, wf[kk], false, xf, (short)0, acc, false, false);
      }
      // C layout (flipped): lane = data row m, VGPR i = channel +i
      int m = rt * 16 + l16;
      bf16x8 h;
#pragma unroll
      for (int i = 0; i < 8; ++i) h[i] = (__bf16)fmaxf(acc[i] + bias[i], 0.f);
      *(u32x4*)&sH[m * LDA + ctile * 16 + hl * 8] = *(const u32x4*)&h;  // 1x b128
    }
  }
  __syncthreads();

  // ---- Phase 2: layer 2  O = relu(H x W2 + b2), computed as O^T tiles ----
  // 4 out-tiles x 10 row-tiles over 8 waves: wave w -> out-tile w/2, row half w&1.
  float* sO = (float*)sA;                         // reuse A region (f32 [160][LDO])
  {
    const int ctile = wave >> 1;                  // out-channel tile 0..3
    const int rbeg  = (wave & 1) * 5;
    bf16x16 wf[4];
    const __bf16* wbase = &sW2T[(ctile * 16 + l16) * LDA];
#pragma unroll
    for (int kk = 0; kk < 4; ++kk) wf[kk] = load_a_frag(wbase, kk, hl);
    f32x8 bias;
#pragma unroll
    for (int i = 0; i < 8; ++i) bias[i] = sb2[ctile * 16 + hl * 8 + i];

    for (int rt = rbeg; rt < rbeg + 5; ++rt) {
      f32x8 acc = {};
      const __bf16* abase = &sH[(rt * 16 + l16) * LDA];
#pragma unroll
      for (int kk = 0; kk < 4; ++kk) {
        bf16x16 xf = load_b_frag(abase, kk, hl);
        acc = __builtin_amdgcn_wmma_f32_16x16x32_bf16(
            false, wf[kk], false, xf, (short)0, acc, false, false);
      }
      int m = rt * 16 + l16;
      f32x8 o;
#pragma unroll
      for (int i = 0; i < 8; ++i) o[i] = fmaxf(acc[i] + bias[i], 0.f);
      float* dst = &sO[m * LDO + ctile * 16 + hl * 8];  // 32B contiguous / lane
      ((u32x4*)dst)[0] = ((const u32x4*)&o)[0];
      ((u32x4*)dst)[1] = ((const u32x4*)&o)[1];
    }
  }
  __syncthreads();

  // ---- Phase 3: mean over 20 neighbors; write rep[occ][hop][0..63] ----
  for (int p = tid; p < OCC_PER_WG * D_OUT; p += 256) {
    int ol = p >> 6, c = p & 63;
    float s = 0.f;
#pragma unroll
    for (int r = 0; r < TOPK; ++r) s += sO[(ol * TOPK + r) * LDO + c];
    int occ = og * OCC_PER_WG + ol;
    rep[(long)occ * (HOPS * D_OUT) + hop * D_OUT + c] = s * (1.0f / TOPK);
  }
}

// Head: u_rep = mean over user's 20 item reps; logit = (u*it) . Wi + bi; sigmoid.
__global__ __launch_bounds__(64) void head_kernel(
    const float* __restrict__ rep, const float* __restrict__ Wi,
    const float* __restrict__ bi, float* __restrict__ out)
{
  __shared__ float red[64];
  const int b = blockIdx.x, t = threadIdx.x;
  float partial = 0.f;
  for (int c = t; c < HOPS * D_OUT; c += 64) {
    float it = rep[(long)b * (HOPS * D_OUT) + c];
    float u = 0.f;
    for (int j = 0; j < IPU; ++j)
      u += rep[(long)(BATCH + b * IPU + j) * (HOPS * D_OUT) + c];
    u *= (1.0f / IPU);
    partial += u * it * Wi[c];
  }
  red[t] = partial;
  __syncthreads();
  for (int s = 32; s > 0; s >>= 1) {
    if (t < s) red[t] += red[t + s];
    __syncthreads();
  }
  if (t == 0) out[b] = 1.0f / (1.0f + expf(-(red[0] + bi[0])));
}

extern "C" void kernel_launch(void* const* d_in, const int* in_sizes, int n_in,
                              void* d_out, int out_size, void* d_ws, size_t ws_size,
                              hipStream_t stream) {
  const int*   item_idxs       = (const int*)d_in[0];
  const int*   user_item_ids   = (const int*)d_in[1];
  const int*   neighbor_ids    = (const int*)d_in[2];
  const float* neighbor_scores = (const float*)d_in[3];
  const float* embed           = (const float*)d_in[4];
  const float* W1              = (const float*)d_in[5];
  const float* b1              = (const float*)d_in[6];
  const float* W2              = (const float*)d_in[7];
  const float* b2              = (const float*)d_in[8];
  const float* Wi              = (const float*)d_in[9];
  const float* bi              = (const float*)d_in[10];

  float* rep = (float*)d_ws;   // [5376][5][64] f32 = 6.9 MB scratch

  dim3 grid(NOCC / OCC_PER_WG, HOPS);   // (672, 5)
  rep_kernel<<<grid, 256, SMEM_BYTES, stream>>>(
      item_idxs, user_item_ids, neighbor_ids, neighbor_scores, embed,
      W1, b1, W2, b2, rep);

  head_kernel<<<BATCH, 64, 0, stream>>>(rep, Wi, bi, (float*)d_out);
}
